// GPT2Model_54271206752756
// MI455X (gfx1250) — matrix operand order repaired and checked by reading, not compiled, not run on hardware
//
#include <hip/hip_runtime.h>
#include <hip/hip_bf16.h>

// ---------------------------------------------------------------------------
// GPT-2 forward for MI455X (gfx1250): all GEMMs + attention via
// v_wmma_f32_16x16x32_f16 (wave32 WMMA), f16 LDS tiles, f32 accumulate.
// B=8, T=1024, D=1024, L=12, H=16, DK=64
// ---------------------------------------------------------------------------

typedef __attribute__((ext_vector_type(8)))  _Float16 v8h;
typedef __attribute__((ext_vector_type(16))) _Float16 v16h;
typedef __attribute__((ext_vector_type(8)))  float    v8f;

#define NB_B 8
#define NB_T 1024
#define NB_D 1024
#define NB_H 16
#define NB_DK 64
#define ROWS (NB_B * NB_T)          // 8192
#define LPITCH 40                   // 32 K-elems + 8 pad; 80B rows, 16B aligned

static __device__ __forceinline__ v8f vzero8() {
  v8f z = {0.f,0.f,0.f,0.f,0.f,0.f,0.f,0.f};
  return z;
}

// A-fragment (16x32 f16, M x K): lane L holds row M=L%16.
// lanes 0-15: k = 0..7 (v0-3) and 16..23 (v4-7); lanes 16-31: k = 8..15, 24..31.
static __device__ __forceinline__ v16h load_frag_A(const _Float16* rowp, int lane) {
  int kb = (lane < 16) ? 0 : 8;
  v8h lo = *(const v8h*)(rowp + kb);
  v8h hi = *(const v8h*)(rowp + kb + 16);
  return __builtin_shufflevector(lo, hi, 0,1,2,3,4,5,6,7,8,9,10,11,12,13,14,15);
}

// B-fragment (32x16 f16, K x N): lane L holds col N=L%16.
// lanes 0-15: k = 0..7 and 16..23; lanes 16-31: k = 8..15, 24..31 (matches A).
static __device__ __forceinline__ v16h load_frag_B(const _Float16* rowp, int lane) {
  int kb = (lane < 16) ? 0 : 8;
  v8h lo = *(const v8h*)(rowp + kb);
  v8h hi = *(const v8h*)(rowp + kb + 16);
  return __builtin_shufflevector(lo, hi, 0,1,2,3,4,5,6,7,8,9,10,11,12,13,14,15);
}

// Branch-free tanh-GELU: tanh(u) = 1 - 2/(1+exp(2u)) on the fast v_exp_f32
// path. Saturates correctly (exp->inf => tanh->1, exp->0 => tanh->-1), no
// exec-mask divergence like libm tanhf.
static __device__ __forceinline__ float gelu_tanh(float x) {
  float u = 0.7978845608f * (x + 0.044715f * x * x * x);
  float e = __expf(2.0f * u);
  float t = 1.0f - 2.0f / (e + 1.0f);
  return 0.5f * x * (1.0f + t);
}

// ---------------------------------------------------------------------------
// h = x + pos_tab (broadcast over batch). 4 floats/thread.
// ---------------------------------------------------------------------------
__global__ __launch_bounds__(256)
void add_pos_kernel(const float* __restrict__ x, const float* __restrict__ pos,
                    float* __restrict__ h) {
  size_t idx = ((size_t)blockIdx.x * 256 + threadIdx.x) * 4;
  size_t rem = idx & ((size_t)NB_T * NB_D - 1);   // T*D is a power of two
  float4 xv = *(const float4*)(x + idx);
  float4 pv = *(const float4*)(pos + rem);
  float4 o;
  o.x = xv.x + pv.x; o.y = xv.y + pv.y; o.z = xv.z + pv.z; o.w = xv.w + pv.w;
  *(float4*)(h + idx) = o;
}

// ---------------------------------------------------------------------------
// Row LayerNorm over D=1024, one block(256) per row.
// ---------------------------------------------------------------------------
__global__ __launch_bounds__(256)
void ln_kernel(const float* __restrict__ x, const float* __restrict__ gamma,
               const float* __restrict__ beta, float* __restrict__ y) {
  __shared__ float sA[256];
  __shared__ float sB[256];
  int t = threadIdx.x;
  const float* xr = x + (size_t)blockIdx.x * NB_D;
  float4 v = ((const float4*)xr)[t];
  sA[t] = v.x + v.y + v.z + v.w;
  sB[t] = v.x*v.x + v.y*v.y + v.z*v.z + v.w*v.w;
  __syncthreads();
  for (int off = 128; off > 0; off >>= 1) {
    if (t < off) { sA[t] += sA[t + off]; sB[t] += sB[t + off]; }
    __syncthreads();
  }
  float mu  = sA[0] * (1.0f / NB_D);
  float var = sB[0] * (1.0f / NB_D) - mu * mu;
  float rs  = rsqrtf(var + 1e-5f);
  float4 g = ((const float4*)gamma)[t];
  float4 b = ((const float4*)beta)[t];
  float4 o;
  o.x = (v.x - mu) * rs * g.x + b.x;
  o.y = (v.y - mu) * rs * g.y + b.y;
  o.z = (v.z - mu) * rs * g.z + b.z;
  o.w = (v.w - mu) * rs * g.w + b.w;
  ((float4*)(y + (size_t)blockIdx.x * NB_D))[t] = o;
}

// ---------------------------------------------------------------------------
// GEMM: C[M,N] = act( A[M,K] * Bw[K,N] + bias ) (+ residual), fp32 in/out,
// f16 WMMA compute. 128x128 block tile, BK=32, 8 waves of 32x64.
// Staging is gather-then-scatter so all 8 global loads stay in flight
// (single loadcnt drain), then packed f16 converts + LDS stores.
// ---------------------------------------------------------------------------
__global__ __launch_bounds__(256)
void gemm_kernel(const float* __restrict__ A, const float* __restrict__ Bw,
                 const float* __restrict__ bias, const float* __restrict__ res,
                 float* __restrict__ C, int M, int N, int K, int act) {
  __shared__ _Float16 As[128][LPITCH];   // [m][k]
  __shared__ _Float16 Bs[128][LPITCH];   // [n][k] (transposed stage)
  int tid = threadIdx.x, lane = tid & 31, wid = tid >> 5;
  int lm = lane & 15;
  int m0 = blockIdx.y * 128, n0 = blockIdx.x * 128;
  int wm = (wid & 3) * 32;       // wave M offset (4 slots)
  int wn = (wid >> 2) * 64;      // wave N offset (2 slots)

  v8f acc[2][4];
#pragma unroll
  for (int i = 0; i < 2; i++)
#pragma unroll
    for (int j = 0; j < 4; j++) acc[i][j] = vzero8();

  int acg = (tid & 7) * 4;       // A stage: 8 float4 per row
  int ar0 = tid >> 3;            // 0..31
  int bcg = (tid & 31) * 4;      // B stage: 32 float4 per row
  int br0 = tid >> 5;            // 0..7

  for (int kt = 0; kt < K; kt += 32) {
    __syncthreads();
    // ---- gather: issue all 8 global loads, keep them all in flight ----
    float4 ta[4], tb[4];
#pragma unroll
    for (int p = 0; p < 4; p++)
      ta[p] = *(const float4*)(A + (size_t)(m0 + ar0 + p * 32) * K + kt + acg);
#pragma unroll
    for (int p = 0; p < 4; p++)
      tb[p] = *(const float4*)(Bw + (size_t)(kt + br0 + p * 8) * N + n0 + bcg);
    // prefetch next k-tile into cache while this tile's WMMAs run
    if (kt + 32 < K) {
      int knext = kt + 32;
#pragma unroll
      for (int p = 0; p < 4; p++)
        __builtin_prefetch(A + (size_t)(m0 + ar0 + p * 32) * K + knext + acg, 0, 3);
#pragma unroll
      for (int p = 0; p < 4; p++)
        __builtin_prefetch(Bw + (size_t)(knext + br0 + p * 8) * N + n0 + bcg, 0, 3);
    }
    // ---- scatter: convert to f16 and store into LDS tiles ----
#pragma unroll
    for (int p = 0; p < 4; p++) {
      int r = ar0 + p * 32;
      As[r][acg + 0] = (_Float16)ta[p].x; As[r][acg + 1] = (_Float16)ta[p].y;
      As[r][acg + 2] = (_Float16)ta[p].z; As[r][acg + 3] = (_Float16)ta[p].w;
    }
#pragma unroll
    for (int p = 0; p < 4; p++) {
      int r = br0 + p * 8;
      Bs[bcg + 0][r] = (_Float16)tb[p].x; Bs[bcg + 1][r] = (_Float16)tb[p].y;
      Bs[bcg + 2][r] = (_Float16)tb[p].z; Bs[bcg + 3][r] = (_Float16)tb[p].w;
    }
    __syncthreads();

    v16h af[2], bf[4];
#pragma unroll
    for (int i = 0; i < 2; i++) af[i] = load_frag_A(&As[wm + i * 16 + lm][0], lane);
#pragma unroll
    for (int j = 0; j < 4; j++) bf[j] = load_frag_B(&Bs[wn + j * 16 + lm][0], lane);
#pragma unroll
    for (int i = 0; i < 2; i++)
#pragma unroll
      for (int j = 0; j < 4; j++)
        acc[i][j] = __builtin_amdgcn_wmma_f32_16x16x32_f16(
            false, af[i], false, bf[j], (short)0, acc[i][j], false, false);
  }

  // epilogue: C layout lane L -> col N=L%16; VGPR v -> row v (+8 for lanes>=16)
  int rsel = (lane < 16) ? 0 : 8;
#pragma unroll
  for (int i = 0; i < 2; i++) {
#pragma unroll
    for (int j = 0; j < 4; j++) {
      int col = n0 + wn + j * 16 + lm;
      float bv = bias[col];
#pragma unroll
      for (int v = 0; v < 8; v++) {
        int row = m0 + wm + i * 16 + v + rsel;
        float xv = acc[i][j][v] + bv;
        if (act) xv = gelu_tanh(xv);
        if (res) xv += res[(size_t)row * N + col];
        C[(size_t)row * N + col] = xv;
      }
    }
  }
}

// ---------------------------------------------------------------------------
// Flash attention: workgroup = (64-query block, b*H). 4 waves, each 16 queries.
// S = (Q/8) K^T via WMMA, causal mask, online softmax, O = P V via WMMA.
// qkv row layout: [q(0..1023) | k(1024..2047) | v(2048..3071)], head = 64 cols.
// ---------------------------------------------------------------------------
__global__ __launch_bounds__(128)
void attn_kernel(const float* __restrict__ qkv, float* __restrict__ aout) {
  __shared__ _Float16 Ks[32][72];       // [key][d]   (72*2 bytes, 16B aligned)
  __shared__ _Float16 Vt[64][LPITCH];   // [d][key]   transposed
  __shared__ _Float16 Ps[4][16][LPITCH];// per-wave P bounce [qrow][key]

  int tid = threadIdx.x, lane = tid & 31, wid = tid >> 5;
  int lm = lane & 15;
  int rsel = (lane < 16) ? 0 : 8;
  int bh = blockIdx.y;
  int bb = bh >> 4;        // batch  (H=16)
  int hh = bh & 15;        // head
  int q0 = blockIdx.x * 64;

  // Q fragments with 1/sqrt(dk)=0.125 folded in (A layout, scattered global)
  int kbA = (lane < 16) ? 0 : 8;
  int qrow = bb * NB_T + q0 + wid * 16 + lm;
  const float* qp = qkv + (size_t)qrow * 3072 + hh * 64;
  v16h qa[2];
#pragma unroll
  for (int ch = 0; ch < 2; ch++) {
    v16h q;
#pragma unroll
    for (int j = 0; j < 8; j++) q[j]     = (_Float16)(qp[ch * 32 + kbA + j] * 0.125f);
#pragma unroll
    for (int j = 0; j < 8; j++) q[8 + j] = (_Float16)(qp[ch * 32 + kbA + 16 + j] * 0.125f);
    qa[ch] = q;
  }

  float mstat[8], lstat[8];
  v8f oacc[4];
#pragma unroll
  for (int v = 0; v < 8; v++) { mstat[v] = -1e30f; lstat[v] = 0.0f; }
#pragma unroll
  for (int dt = 0; dt < 4; dt++) oacc[dt] = vzero8();

  int ntiles = (q0 + 64) >> 5;   // key tiles of 32, up to and incl. this q-block
  for (int jt = 0; jt < ntiles; jt++) {
    int k0 = jt * 32;
    __syncthreads();
    { // stage K [32][64] and V transposed [64][32]: gather then scatter
      int cg = (tid & 15) * 4;
      int r0 = tid >> 4;                // 0..7
      float4 tk[4], tv[4];
#pragma unroll
      for (int p = 0; p < 4; p++) {
        const float* src = qkv + (size_t)(bb * NB_T + k0 + r0 + p * 8) * 3072
                           + 1024 + hh * 64 + cg;
        tk[p] = *(const float4*)src;
        tv[p] = *(const float4*)(src + 1024);
      }
#pragma unroll
      for (int p = 0; p < 4; p++) {
        int r = r0 + p * 8;
        Ks[r][cg + 0] = (_Float16)tk[p].x; Ks[r][cg + 1] = (_Float16)tk[p].y;
        Ks[r][cg + 2] = (_Float16)tk[p].z; Ks[r][cg + 3] = (_Float16)tk[p].w;
      }
#pragma unroll
      for (int p = 0; p < 4; p++) {
        int r = r0 + p * 8;
        Vt[cg + 0][r] = (_Float16)tv[p].x; Vt[cg + 1][r] = (_Float16)tv[p].y;
        Vt[cg + 2][r] = (_Float16)tv[p].z; Vt[cg + 3][r] = (_Float16)tv[p].w;
      }
    }
    __syncthreads();

    // S = Q K^T for two 16-key subtiles
    v8f sfr[2];
#pragma unroll
    for (int st = 0; st < 2; st++) {
      v8f s = vzero8();
#pragma unroll
      for (int ch = 0; ch < 2; ch++) {
        v16h kf = load_frag_B(&Ks[st * 16 + lm][ch * 32], lane);
        s = __builtin_amdgcn_wmma_f32_16x16x32_f16(
            false, qa[ch], false, kf, (short)0, s, false, false);
      }
      sfr[st] = s;
    }

    // causal mask + online softmax (row = v+rsel, col = key in 16-lane half)
#pragma unroll
    for (int v = 0; v < 8; v++) {
      int qi = q0 + wid * 16 + v + rsel;
      float s0 = (k0 + lm      <= qi) ? sfr[0][v] : -1e10f;
      float s1 = (k0 + 16 + lm <= qi) ? sfr[1][v] : -1e10f;
      float rm = fmaxf(s0, s1);
      rm = fmaxf(rm, __shfl_xor(rm, 1));
      rm = fmaxf(rm, __shfl_xor(rm, 2));
      rm = fmaxf(rm, __shfl_xor(rm, 4));
      rm = fmaxf(rm, __shfl_xor(rm, 8));
      float mnew = fmaxf(mstat[v], rm);
      float corr = __expf(mstat[v] - mnew);
      float p0 = __expf(s0 - mnew);
      float p1 = __expf(s1 - mnew);
      float ps = p0 + p1;
      ps += __shfl_xor(ps, 1);
      ps += __shfl_xor(ps, 2);
      ps += __shfl_xor(ps, 4);
      ps += __shfl_xor(ps, 8);
      lstat[v] = lstat[v] * corr + ps;
      mstat[v] = mnew;
#pragma unroll
      for (int dt = 0; dt < 4; dt++) oacc[dt][v] *= corr;
      Ps[wid][v + rsel][lm]      = (_Float16)p0;
      Ps[wid][v + rsel][16 + lm] = (_Float16)p1;
    }
    asm volatile("s_wait_dscnt 0" ::: "memory");  // own-wave P write -> read

    // O += P V  (P as A-frag from LDS bounce, V as B-frags, 4 d-tiles)
    v16h pa = load_frag_A(&Ps[wid][lm][0], lane);
#pragma unroll
    for (int dt = 0; dt < 4; dt++) {
      v16h vf = load_frag_B(&Vt[dt * 16 + lm][0], lane);
      oacc[dt] = __builtin_amdgcn_wmma_f32_16x16x32_f16(
          false, pa, false, vf, (short)0, oacc[dt], false, false);
    }
  }

  // normalize and write merged-head output
#pragma unroll
  for (int v = 0; v < 8; v++) {
    float inv = 1.0f / lstat[v];
    int row = bb * NB_T + q0 + wid * 16 + v + rsel;
    float* op = aout + (size_t)row * (NB_H * NB_DK) + hh * 64;
#pragma unroll
    for (int dt = 0; dt < 4; dt++) op[dt * 16 + lm] = oacc[dt][v] * inv;
  }
}

// ---------------------------------------------------------------------------
// Orchestration
// ---------------------------------------------------------------------------
extern "C" void kernel_launch(void* const* d_in, const int* in_sizes, int n_in,
                              void* d_out, int out_size, void* d_ws, size_t ws_size,
                              hipStream_t stream) {
  (void)in_sizes; (void)n_in; (void)out_size; (void)ws_size;
  const float* x      = (const float*)d_in[0];
  const float* pos    = (const float*)d_in[1];
  const float* ln1_g  = (const float*)d_in[2];
  const float* ln1_b  = (const float*)d_in[3];
  const float* attn_w = (const float*)d_in[4];
  const float* attn_b = (const float*)d_in[5];
  const float* proj_w = (const float*)d_in[6];
  const float* proj_b = (const float*)d_in[7];
  const float* ln2_g  = (const float*)d_in[8];
  const float* ln2_b  = (const float*)d_in[9];
  const float* fc_w   = (const float*)d_in[10];
  const float* fc_b   = (const float*)d_in[11];
  const float* fc2_w  = (const float*)d_in[12];
  const float* fc2_b  = (const float*)d_in[13];
  const float* lnf_g  = (const float*)d_in[14];
  const float* lnf_b  = (const float*)d_in[15];
  float* out = (float*)d_out;

  // workspace: h (32MB) | act (32MB, ln-out / attn-out) | big (128MB, qkv / mid)
  float* h   = (float*)d_ws;
  float* act = (float*)((char*)d_ws + (size_t)(32u << 20));
  float* big = (float*)((char*)d_ws + (size_t)(64u << 20));

  add_pos_kernel<<<ROWS * NB_D / (256 * 4), 256, 0, stream>>>(x, pos, h);

  for (int i = 0; i < 12; i++) {
    // a_in = LN1(h)
    ln_kernel<<<ROWS, 256, 0, stream>>>(h, ln1_g + i * NB_D, ln1_b + i * NB_D, act);
    // qkv = a_in @ attn_w + attn_b            (8192 x 3072, K=1024)
    gemm_kernel<<<dim3(3072 / 128, ROWS / 128), 256, 0, stream>>>(
        act, attn_w + (size_t)i * NB_D * 3072, attn_b + (size_t)i * 3072,
        nullptr, big, ROWS, 3072, NB_D, 0);
    // attention -> act
    attn_kernel<<<dim3(NB_T / 64, NB_B * NB_H), 128, 0, stream>>>(big, act);
    // h = h + act @ proj_w + proj_b           (8192 x 1024, K=1024)
    gemm_kernel<<<dim3(NB_D / 128, ROWS / 128), 256, 0, stream>>>(
        act, proj_w + (size_t)i * NB_D * NB_D, proj_b + (size_t)i * NB_D,
        h, h, ROWS, NB_D, NB_D, 0);
    // m_in = LN2(h)
    ln_kernel<<<ROWS, 256, 0, stream>>>(h, ln2_g + i * NB_D, ln2_b + i * NB_D, act);
    // mid = gelu(m_in @ fc_w + fc_b)          (8192 x 4096, K=1024)
    gemm_kernel<<<dim3(4096 / 128, ROWS / 128), 256, 0, stream>>>(
        act, fc_w + (size_t)i * NB_D * 4096, fc_b + (size_t)i * 4096,
        nullptr, big, ROWS, 4096, NB_D, 1);
    // h = h + mid @ fc2_w + fc2_b             (8192 x 1024, K=4096)
    gemm_kernel<<<dim3(NB_D / 128, ROWS / 128), 256, 0, stream>>>(
        big, fc2_w + (size_t)i * 4096 * NB_D, fc2_b + (size_t)i * NB_D,
        h, h, ROWS, NB_D, 4096, 0);
  }

  ln_kernel<<<ROWS, 256, 0, stream>>>(h, lnf_g, lnf_b, out);
}
